// CrossCovarianceAttn_12498354831860
// MI455X (gfx1250) — matrix operand order repaired and checked by reading, not compiled, not run on hardware
//
#include <hip/hip_runtime.h>
#include <hip/hip_bf16.h>
#include <cmath>

typedef __attribute__((ext_vector_type(16))) _Float16 v16h;
typedef __attribute__((ext_vector_type(8)))  _Float16 v8h;
typedef __attribute__((ext_vector_type(8)))  float    v8f;

// ---------------------------------------------------------------------------
// WMMA fragment loaders (wave32), contiguous k order in A and B.
// A (16x32 f16, MxK): lanes 0-15 -> M=lane, K=0..15;
//                     lanes 16-31 -> M=lane-16, K=16..31.
// B (32x16 f16, KxN): lanes 0-15 -> N=lane, K=0..15 (v index);
//                     lanes 16-31 -> N=lane-16, K=16..31.
// C/D (16x16 f32):    lanes 0-15 -> M=r, N=lane; lanes 16-31 -> M=r+8, N=lane-16.
// ---------------------------------------------------------------------------
__device__ __forceinline__ v16h frag_a16(const _Float16* p, int lda) {
  const int lane = threadIdx.x & 31;
  const int m = lane & 15;
  const int kb = (lane < 16) ? 0 : 16;
  const _Float16* row = p + m * lda + kb;
  v8h lo = *(const v8h*)(row);        // K = kb .. kb+7
  v8h hi = *(const v8h*)(row + 8);    // K = kb+8 .. kb+15
  v16h r;
#pragma unroll
  for (int v = 0; v < 8; ++v) { r[v] = lo[v]; r[v + 8] = hi[v]; }
  return r;
}

__device__ __forceinline__ v16h frag_b16s(const _Float16* p, int ldb) {
  // Strided gather fallback: element (k, n) = p[k*ldb + n]
  const int lane = threadIdx.x & 31;
  const int n = lane & 15;
  const int kb = (lane < 16) ? 0 : 16;
  v16h r;
#pragma unroll
  for (int v = 0; v < 16; ++v) r[v] = p[(kb + v) * ldb + n];
  return r;
}

__device__ __forceinline__ v16h frag_bT16(const _Float16* p, int ld) {
  // B supplied as N x K row-major: element (k, n) = p[n*ld + k].
  // 32B contiguous and 32B aligned per lane -> single v16h load.
  const int lane = threadIdx.x & 31;
  const int n = lane & 15;
  const int kb = (lane < 16) ? 0 : 16;
  return *(const v16h*)(p + n * ld + kb);
}

// ---------------------------------------------------------------------------
// Kernel 0a: format weight matrix (K=768 rows x N cols, f32 row-major) into
// fragment-ready f16 layout:
//   Wf[(((nt*24)+kt)*32 + flane)*16 + v],
//   k = kt*32 + (flane>=16 ? 16:0) + v,  n = nt*16 + (flane&15)
// ---------------------------------------------------------------------------
__global__ void __launch_bounds__(256) wprep_kernel(
    const float* __restrict__ W, _Float16* __restrict__ Wf, int N) {
  const int idx = blockIdx.x * 256 + threadIdx.x;
  const int flane = idx & 31;
  const int kt = (idx >> 5) % 24;
  const int nt = (idx >> 5) / 24;
  const int n = (nt << 4) + (flane & 15);
  const int kbase = (kt << 5) + ((flane >> 4) << 4);
  v16h r;
#pragma unroll
  for (int v = 0; v < 16; ++v)
    r[v] = (_Float16)W[(size_t)(kbase + v) * N + n];
  *(v16h*)(Wf + (size_t)idx * 16) = r;
}

// ---------------------------------------------------------------------------
// Kernel 0b: x (f32) -> xh (f16), fully coalesced. One v8h per thread.
// ---------------------------------------------------------------------------
__global__ void __launch_bounds__(256) xprep_kernel(
    const float* __restrict__ x, _Float16* __restrict__ xh) {
  const size_t i = ((size_t)blockIdx.x * 256 + threadIdx.x) * 8;
  float4 f0 = *(const float4*)(x + i);
  float4 f1 = *(const float4*)(x + i + 4);
  v8h h;
  h[0] = (_Float16)f0.x; h[1] = (_Float16)f0.y;
  h[2] = (_Float16)f0.z; h[3] = (_Float16)f0.w;
  h[4] = (_Float16)f1.x; h[5] = (_Float16)f1.y;
  h[6] = (_Float16)f1.z; h[7] = (_Float16)f1.w;
  *(v8h*)(xh + i) = h;
}

// ---------------------------------------------------------------------------
// Kernel 1: qkv = x @ W_qkv  (M=32768, N=2304, K=768), 128x128 block tile.
// 8 waves as 4(M)x2(N); wave tile 32x64 = 2x4 WMMA tiles.
// Output scattered into q/k/v (b,h,d,n) f16 layout, 16B per-lane stores.
// ---------------------------------------------------------------------------
__global__ void __launch_bounds__(256) qkv_gemm_kernel(
    const _Float16* __restrict__ xh, const _Float16* __restrict__ Wf,
    _Float16* __restrict__ qT, _Float16* __restrict__ kT,
    _Float16* __restrict__ vT) {
  __shared__ __align__(32) _Float16 aL[128 * 32];
  __shared__ __align__(32) _Float16 bL[8 * 512];
  const int tid = threadIdx.x;
  const int j0 = blockIdx.x * 128;
  const int nt0 = j0 >> 4;
  const int m0 = blockIdx.y * 128;
  const int w = tid >> 5, lane = tid & 31;
  const int wm = w & 3, wn = w >> 2;

  v8f acc[2][4] = {};
  for (int kt = 0; kt < 24; ++kt) {
    const int k0 = kt << 5;
    // A tile 128x32 f16: straight-line, 2 x b128 per thread + prefetch
#pragma unroll
    for (int it = 0; it < 2; ++it) {
      const int i = tid + it * 256;
      const int row = i >> 2, c = (i & 3) << 3;
      const _Float16* src = xh + (size_t)(m0 + row) * 768 + k0 + c;
      *(v8h*)(aL + row * 32 + c) = *(const v8h*)src;
      __builtin_prefetch(src + 32, 0, 3);   // speculative: OOB is dropped
    }
    // B: 8 fragment-ready tiles of 512 halves: 2 x b128 per thread
#pragma unroll
    for (int it = 0; it < 2; ++it) {
      const int i = tid + it * 256;
      const int tile = i >> 6, off = (i & 63) << 3;
      *(v8h*)(bL + tile * 512 + off) =
          *(const v8h*)(Wf + (((size_t)(nt0 + tile) * 24 + kt) << 9) + off);
    }
    __syncthreads();
    v16h bfr[4];
#pragma unroll
    for (int tn = 0; tn < 4; ++tn)
      bfr[tn] = *(const v16h*)(bL + (wn * 4 + tn) * 512 + lane * 16);
#pragma unroll
    for (int tm = 0; tm < 2; ++tm) {
      v16h a = frag_a16(aL + (wm * 32 + tm * 16) * 32, 32);
#pragma unroll
      for (int tn = 0; tn < 4; ++tn)
        acc[tm][tn] = __builtin_amdgcn_wmma_f32_16x16x32_f16(
            false, a, false, bfr[tn], (short)0, acc[tm][tn], false, false);
    }
    __syncthreads();
  }

  const int bb = m0 >> 12;                 // 4096 tokens per batch
  const int hi8 = (lane >> 4) << 3;
  const int ncol = lane & 15;
#pragma unroll
  for (int tm = 0; tm < 2; ++tm) {
#pragma unroll
    for (int tn = 0; tn < 4; ++tn) {
      const int jg = j0 + wn * 64 + tn * 16 + ncol;
      const int which = jg / 768;
      const int rem = jg % 768;
      const int hh = rem / 96, dd = rem % 96;
      _Float16* dst = (which == 0) ? qT : (which == 1) ? kT : vT;
      const size_t base = ((size_t)((bb * 8 + hh) * 96 + dd)) << 12;
      const int nn = (m0 & 4095) + wm * 32 + tm * 16 + hi8;
      v8h hv;
#pragma unroll
      for (int r = 0; r < 8; ++r) hv[r] = (_Float16)acc[tm][tn][r];
      *(v8h*)(dst + base + nn) = hv;       // contiguous 16B per lane
    }
  }
}

// ---------------------------------------------------------------------------
// Kernel 2: inverse L2 norms of q and k rows over n (4096, contiguous).
// ---------------------------------------------------------------------------
__global__ void __launch_bounds__(256) norm_kernel(
    const _Float16* __restrict__ qT, const _Float16* __restrict__ kT,
    float* __restrict__ invq, float* __restrict__ invk) {
  __shared__ float red[256];
  int row = blockIdx.x;
  const bool isK = row >= 6144;
  if (isK) row -= 6144;
  const _Float16* src = (isK ? kT : qT) + (size_t)row * 4096;
  float s = 0.f;
#pragma unroll
  for (int it = 0; it < 2; ++it) {
    const int i = threadIdx.x + it * 256;
    v8h h = *(const v8h*)(src + i * 8);
#pragma unroll
    for (int j = 0; j < 8; ++j) { const float v = (float)h[j]; s += v * v; }
  }
  red[threadIdx.x] = s;
  __syncthreads();
  for (int o = 128; o > 0; o >>= 1) {
    if (threadIdx.x < o) red[threadIdx.x] += red[threadIdx.x + o];
    __syncthreads();
  }
  if (threadIdx.x == 0) {
    const float inv = 1.0f / fmaxf(sqrtf(red[0]), 1e-12f);
    (isK ? invk : invq)[row] = inv;
  }
}

// ---------------------------------------------------------------------------
// Kernel 3: per (b,h): S = Q.K^T (96x96 over K=4096), scale by
// invq[d]*invk[e]*temp[h], softmax over e, store f16 attn.
// 6 waves, wave w owns rows [16w, 16w+16). K-chunk 64 (2 WMMA steps/stage).
// ---------------------------------------------------------------------------
__global__ void __launch_bounds__(192) attn_kernel(
    const _Float16* __restrict__ qT, const _Float16* __restrict__ kT,
    const float* __restrict__ invq, const float* __restrict__ invk,
    const float* __restrict__ temp, _Float16* __restrict__ attnOut) {
  __shared__ __align__(32) _Float16 qL[96 * 64];
  __shared__ __align__(32) _Float16 kL[96 * 64];
  __shared__ float sL[96 * 100];
  const int bh = blockIdx.x;
  const int hh = bh & 7;
  const int tid = threadIdx.x;
  const int w = tid >> 5;
  const int lane = tid & 31;
  const size_t rowbase = (size_t)bh * 96;

  v8f acc[6] = {};
  for (int k0 = 0; k0 < 4096; k0 += 64) {
#pragma unroll
    for (int it = 0; it < 4; ++it) {
      const int i = tid + it * 192;
      const int row = i >> 3, c = (i & 7) << 3;
      *(v8h*)(qL + row * 64 + c) =
          *(const v8h*)(qT + (rowbase + row) * 4096 + k0 + c);
      *(v8h*)(kL + row * 64 + c) =
          *(const v8h*)(kT + (rowbase + row) * 4096 + k0 + c);
    }
    __syncthreads();
#pragma unroll
    for (int kc = 0; kc < 2; ++kc) {
      v16h a = frag_a16(qL + w * 16 * 64 + kc * 32, 64);
#pragma unroll
      for (int e = 0; e < 6; ++e) {
        v16h b = frag_bT16(kL + e * 16 * 64 + kc * 32, 64);
        acc[e] = __builtin_amdgcn_wmma_f32_16x16x32_f16(
            false, a, false, b, (short)0, acc[e], false, false);
      }
    }
    __syncthreads();
  }

  const int hi8 = (lane >> 4) << 3, ncol = lane & 15;
#pragma unroll
  for (int e = 0; e < 6; ++e)
#pragma unroll
    for (int r = 0; r < 8; ++r)
      sL[(w * 16 + hi8 + r) * 100 + e * 16 + ncol] = acc[e][r];
  __syncthreads();

  if (tid < 96) {
    const float iq = invq[rowbase + tid];
    const float t = temp[hh];
    float mx = -1e30f;
    for (int e = 0; e < 96; ++e) {
      const float v = sL[tid * 100 + e] * iq * invk[rowbase + e] * t;
      mx = fmaxf(mx, v);
    }
    float sum = 0.f;
    for (int e = 0; e < 96; ++e) {
      const float v = sL[tid * 100 + e] * iq * invk[rowbase + e] * t;
      sum += __expf(v - mx);
    }
    const float is = 1.0f / sum;
    for (int e = 0; e < 96; ++e) {
      const float v = sL[tid * 100 + e] * iq * invk[rowbase + e] * t;
      attnOut[(rowbase + tid) * 96 + e] = (_Float16)(__expf(v - mx) * is);
    }
  }
}

// ---------------------------------------------------------------------------
// Kernel 4: per (b,h, n-chunk of 128): out = attn(96x96) @ v(96x128),
// scattered into y (b,n,c) f16. 8 waves; wave w owns cols [16w,16w+16).
// ---------------------------------------------------------------------------
__global__ void __launch_bounds__(256) av_kernel(
    const _Float16* __restrict__ attn, const _Float16* __restrict__ vT,
    _Float16* __restrict__ y) {
  __shared__ __align__(32) _Float16 aL[96 * 96];
  __shared__ __align__(32) _Float16 vL[96 * 128];
  const int bh = blockIdx.x, n0 = blockIdx.y * 128;
  const int bb = bh >> 3, hh = bh & 7;
  const int tid = threadIdx.x, w = tid >> 5, lane = tid & 31;

  for (int i = tid; i < 1152; i += 256) {
    const int row = i / 12, c = (i % 12) << 3;
    *(v8h*)(aL + row * 96 + c) =
        *(const v8h*)(attn + ((size_t)bh * 96 + row) * 96 + c);
  }
#pragma unroll
  for (int it = 0; it < 6; ++it) {
    const int i = tid + it * 256;
    const int row = i >> 4, c = (i & 15) << 3;
    *(v8h*)(vL + row * 128 + c) =
        *(const v8h*)(vT + ((size_t)bh * 96 + row) * 4096 + n0 + c);
  }
  __syncthreads();

  v8f acc[6] = {};
#pragma unroll
  for (int kc = 0; kc < 3; ++kc) {
    v16h b = frag_b16s(vL + kc * 32 * 128 + w * 16, 128);
#pragma unroll
    for (int mt = 0; mt < 6; ++mt) {
      v16h a = frag_a16(aL + mt * 16 * 96 + kc * 32, 96);
      acc[mt] = __builtin_amdgcn_wmma_f32_16x16x32_f16(
          false, a, false, b, (short)0, acc[mt], false, false);
    }
  }

  const int hi8 = (lane >> 4) << 3, ncol = lane & 15;
  const int nn = n0 + w * 16 + ncol;
  const size_t rowoff = ((size_t)bb * 4096 + nn) * 768 + hh * 96;
#pragma unroll
  for (int mt = 0; mt < 6; ++mt) {
    v8h hv;
#pragma unroll
    for (int r = 0; r < 8; ++r) hv[r] = (_Float16)acc[mt][r];
    *(v8h*)(y + rowoff + mt * 16 + hi8) = hv;   // contiguous 16B per lane
  }
}

// ---------------------------------------------------------------------------
// Kernel 5: out = y @ W_proj + b_proj  (M=32768, N=768, K=768), fp32 out.
// Same 128x128 tiling; B pre-formatted fragment-ready.
// ---------------------------------------------------------------------------
__global__ void __launch_bounds__(256) proj_gemm_kernel(
    const _Float16* __restrict__ y, const _Float16* __restrict__ Wf,
    const float* __restrict__ bias, float* __restrict__ out) {
  __shared__ __align__(32) _Float16 aL[128 * 32];
  __shared__ __align__(32) _Float16 bL[8 * 512];
  const int tid = threadIdx.x;
  const int j0 = blockIdx.x * 128, m0 = blockIdx.y * 128;
  const int nt0 = j0 >> 4;
  const int w = tid >> 5, lane = tid & 31;
  const int wm = w & 3, wn = w >> 2;

  v8f acc[2][4] = {};
  for (int kt = 0; kt < 24; ++kt) {
    const int k0 = kt << 5;
#pragma unroll
    for (int it = 0; it < 2; ++it) {
      const int i = tid + it * 256;
      const int row = i >> 2, c = (i & 3) << 3;
      const _Float16* src = y + (size_t)(m0 + row) * 768 + k0 + c;
      *(v8h*)(aL + row * 32 + c) = *(const v8h*)src;
      __builtin_prefetch(src + 32, 0, 3);
    }
#pragma unroll
    for (int it = 0; it < 2; ++it) {
      const int i = tid + it * 256;
      const int tile = i >> 6, off = (i & 63) << 3;
      *(v8h*)(bL + tile * 512 + off) =
          *(const v8h*)(Wf + (((size_t)(nt0 + tile) * 24 + kt) << 9) + off);
    }
    __syncthreads();
    v16h bfr[4];
#pragma unroll
    for (int tn = 0; tn < 4; ++tn)
      bfr[tn] = *(const v16h*)(bL + (wn * 4 + tn) * 512 + lane * 16);
#pragma unroll
    for (int tm = 0; tm < 2; ++tm) {
      v16h a = frag_a16(aL + (wm * 32 + tm * 16) * 32, 32);
#pragma unroll
      for (int tn = 0; tn < 4; ++tn)
        acc[tm][tn] = __builtin_amdgcn_wmma_f32_16x16x32_f16(
            false, a, false, bfr[tn], (short)0, acc[tm][tn], false, false);
    }
    __syncthreads();
  }

  const int hi8 = (lane >> 4) << 3, ncol = lane & 15;
#pragma unroll
  for (int tm = 0; tm < 2; ++tm) {
#pragma unroll
    for (int tn = 0; tn < 4; ++tn) {
      const int jg = j0 + wn * 64 + tn * 16 + ncol;
      const float bv = bias[jg];
      const int mg = m0 + wm * 32 + tm * 16 + hi8;
#pragma unroll
      for (int r = 0; r < 8; ++r)
        out[(size_t)(mg + r) * 768 + jg] = acc[tm][tn][r] + bv;
    }
  }
}

// ---------------------------------------------------------------------------
extern "C" void kernel_launch(void* const* d_in, const int* in_sizes, int n_in,
                              void* d_out, int out_size, void* d_ws,
                              size_t ws_size, hipStream_t stream) {
  const float* x     = (const float*)d_in[0];
  const float* Wqkv  = (const float*)d_in[1];
  const float* Wproj = (const float*)d_in[2];
  const float* bproj = (const float*)d_in[3];
  const float* temp  = (const float*)d_in[4];
  float* out = (float*)d_out;

  const size_t R  = 6144;          // b*h*d rows
  const size_t QE = R * 4096;      // elements per q/k/v tensor
  _Float16* qT     = (_Float16*)d_ws;
  _Float16* kT     = qT + QE;
  _Float16* vT     = kT + QE;
  float*    invq   = (float*)(vT + QE);
  float*    invk   = invq + R;
  _Float16* attn   = (_Float16*)(invk + R);            // 64*96*96
  _Float16* y      = attn + (size_t)64 * 96 * 96;      // 32768*768
  _Float16* xh     = y;   // alias: xh dead before av_kernel writes y
  _Float16* Wqkvf  = y + (size_t)32768 * 768;          // 768*2304 frag-ready
  _Float16* Wprojf = Wqkvf + (size_t)768 * 2304;       // 768*768 frag-ready

  wprep_kernel<<<dim3(432), 256, 0, stream>>>(Wqkv, Wqkvf, 2304);
  wprep_kernel<<<dim3(144), 256, 0, stream>>>(Wproj, Wprojf, 768);
  xprep_kernel<<<dim3(12288), 256, 0, stream>>>(x, xh);
  qkv_gemm_kernel<<<dim3(18, 256), 256, 0, stream>>>(xh, Wqkvf, qT, kT, vT);
  norm_kernel<<<dim3(12288), 256, 0, stream>>>(qT, kT, invq, invk);
  attn_kernel<<<dim3(64), 192, 0, stream>>>(qT, kT, invq, invk, temp, attn);
  av_kernel<<<dim3(64, 32), 256, 0, stream>>>(attn, vT, y);
  proj_gemm_kernel<<<dim3(6, 256), 256, 0, stream>>>(y, Wprojf, bproj, out);
}
